// VSN_1228360646983
// MI455X (gfx1250) — matrix operand order repaired and checked by reading, not compiled, not run on hardware
//
#include <hip/hip_runtime.h>

typedef __attribute__((ext_vector_type(16))) _Float16 v16h;
typedef __attribute__((ext_vector_type(8)))  float    v8f;

#define BT    16384   // B*T
#define NBLK  65      // 64 expert K-blocks (K=64 each) + 1 bias block
#define TM    64      // rows per workgroup
#define LDF   68      // padded LDS stride for 64-wide rows (272B = 17*16B)
#define LDE   36      // padded LDS stride for 32-wide rows (144B = 9*16B)

__device__ __forceinline__ float eluf(float z) {
  return z > 0.0f ? z : (__expf(z) - 1.0f);
}

#define WMMA_F16(A, B, C) \
  __builtin_amdgcn_wmma_f32_16x16x32_f16(false, (A), false, (B), (short)0, (C), false, false)

// Build f16 A-operand (16x32 lane layout: elems 0..7 <- p[0..7], 8..15 <- p[16..23])
__device__ __forceinline__ v16h pack_a(const float* p) {
  float4 q0 = *(const float4*)(p);
  float4 q1 = *(const float4*)(p + 4);
  float4 q2 = *(const float4*)(p + 16);
  float4 q3 = *(const float4*)(p + 20);
  v16h v;
  v[0]=(_Float16)q0.x; v[1]=(_Float16)q0.y; v[2]=(_Float16)q0.z; v[3]=(_Float16)q0.w;
  v[4]=(_Float16)q1.x; v[5]=(_Float16)q1.y; v[6]=(_Float16)q1.z; v[7]=(_Float16)q1.w;
  v[8]=(_Float16)q2.x; v[9]=(_Float16)q2.y; v[10]=(_Float16)q2.z; v[11]=(_Float16)q2.w;
  v[12]=(_Float16)q3.x; v[13]=(_Float16)q3.y; v[14]=(_Float16)q3.z; v[15]=(_Float16)q3.w;
  return v;
}

// Build f16 B-operand (32x16 lane layout: elems 0..7 <- p[0..7], 8..15 <- p[16..23])
__device__ __forceinline__ v16h pack_b(const float* p) {
  float4 q0 = *(const float4*)(p);
  float4 q1 = *(const float4*)(p + 4);
  float4 q2 = *(const float4*)(p + 16);
  float4 q3 = *(const float4*)(p + 20);
  v16h v;
  v[0]=(_Float16)q0.x; v[1]=(_Float16)q0.y; v[2]=(_Float16)q0.z; v[3]=(_Float16)q0.w;
  v[4]=(_Float16)q1.x; v[5]=(_Float16)q1.y; v[6]=(_Float16)q1.z; v[7]=(_Float16)q1.w;
  v[8]=(_Float16)q2.x; v[9]=(_Float16)q2.y; v[10]=(_Float16)q2.z; v[11]=(_Float16)q2.w;
  v[12]=(_Float16)q3.x; v[13]=(_Float16)q3.y; v[14]=(_Float16)q3.z; v[15]=(_Float16)q3.w;
  return v;
}

// A-operand for the fused expert GEMM: elu(x*w1 + b1) * w_t, built in registers.
__device__ __forceinline__ v16h make_hw(const float* we, const float* be,
                                        float xv, float wt) {
  float4 w0 = *(const float4*)(we);
  float4 w1 = *(const float4*)(we + 4);
  float4 w2 = *(const float4*)(we + 16);
  float4 w3 = *(const float4*)(we + 20);
  float4 b0 = *(const float4*)(be);
  float4 b1 = *(const float4*)(be + 4);
  float4 b2 = *(const float4*)(be + 16);
  float4 b3 = *(const float4*)(be + 20);
  float wf[16] = {w0.x,w0.y,w0.z,w0.w, w1.x,w1.y,w1.z,w1.w,
                  w2.x,w2.y,w2.z,w2.w, w3.x,w3.y,w3.z,w3.w};
  float bf[16] = {b0.x,b0.y,b0.z,b0.w, b1.x,b1.y,b1.z,b1.w,
                  b2.x,b2.y,b2.z,b2.w, b3.x,b3.y,b3.z,b3.w};
  v16h v;
#pragma unroll
  for (int e = 0; e < 16; ++e)
    v[e] = (_Float16)(eluf(fmaf(xv, wf[e], bf[e])) * wt);
  return v;
}

// ---------------------------------------------------------------------------
// Prep: pack [exp_w3 ; exp_b3] into per-lane WMMA B-operand layout (f16).
// chunkIdx = (fb*4 + nt)*2 + kh ; element = chunk*512 + l*16 + e
// lane l: n = nt*16 + (l&15), K = kh*32 + (l>>4)*8 + (e<8 ? e : 8+e)
// ---------------------------------------------------------------------------
__global__ void vsn_prep_B(const float* __restrict__ exp_w3,
                           const float* __restrict__ exp_b3,
                           _Float16* __restrict__ Bws) {
  int idx = blockIdx.x * blockDim.x + threadIdx.x;
  if (idx >= NBLK * 4096) return;
  int e     = idx & 15;
  int l     = (idx >> 4) & 31;
  int chunk = idx >> 9;
  int kh = chunk & 1;
  int nt = (chunk >> 1) & 3;
  int fb = chunk >> 3;
  int k = kh * 32 + (l >> 4) * 8 + (e < 8 ? e : 8 + e);
  int n = nt * 16 + (l & 15);
  float v = (fb < 64) ? exp_w3[((size_t)fb * 64 + k) * 64 + n]
                      : exp_b3[(size_t)k * 64 + n];
  Bws[idx] = (_Float16)v;
}

// ---------------------------------------------------------------------------
// Fused kernel: gating FFN (WMMA) -> softmax -> expert GEMM (WMMA, K-split)
// ---------------------------------------------------------------------------
__global__ __launch_bounds__(256)
void vsn_main(const float* __restrict__ x, const float* __restrict__ s,
              const float* __restrict__ ffn_w1, const float* __restrict__ ffn_b1,
              const float* __restrict__ ffn_w2, const float* __restrict__ ffn_b2,
              const float* __restrict__ ffn_w3, const float* __restrict__ ffn_b3,
              const float* __restrict__ exp_w1, const float* __restrict__ exp_b1,
              const _Float16* __restrict__ Bws, float* __restrict__ out) {
  __shared__ __align__(16) float xs [64 * LDF];  // x tile        [row][f]
  __shared__ __align__(16) float ss [64 * LDE];  // s tile        [row][e]
  __shared__ __align__(16) float w1t[64 * LDF];  // ffn_w1^T      [h][f]
  __shared__ __align__(16) float w2t[64 * LDE];  // ffn_w2^T      [h][e]
  __shared__ __align__(16) float w3t[64 * LDF];  // ffn_w3^T      [f][h]
  __shared__ __align__(16) float ew1[64 * LDF];  // exp_w1        [f][h]
  __shared__ __align__(16) float eb1[64 * LDF];  // exp_b1        [f][h]
  __shared__ __align__(16) float as_[64 * LDF];  // elu(g), later partial-sum buf
  __shared__ __align__(16) float wts[64 * LDF];  // softmax wts   [row][f]
  __shared__ float bs12[64], b3s[64];

  const int tid = threadIdx.x;
  const int row_base = blockIdx.x * TM;

  // ---- stage LDS (weights transposed for row-contiguous B operands) ----
  for (int i = tid; i < 64 * 64; i += 256) {
    int r = i >> 6, c = i & 63;
    xs [r * LDF + c] = x[(size_t)(row_base + r) * 64 + c];
    w1t[c * LDF + r] = ffn_w1[i];   // [f][h] -> [h][f]
    w3t[c * LDF + r] = ffn_w3[i];   // [h][f] -> [f][h]
    ew1[r * LDF + c] = exp_w1[i];
    eb1[r * LDF + c] = exp_b1[i];
  }
  for (int i = tid; i < 64 * 32; i += 256) {
    int r = i >> 5, c = i & 31;
    ss[r * LDE + c] = s[(size_t)(row_base + r) * 32 + c];
  }
  for (int i = tid; i < 32 * 64; i += 256) {
    int r = i >> 6, c = i & 63;      // r = e, c = h
    w2t[c * LDE + r] = ffn_w2[i];    // [e][h] -> [h][e]
  }
  if (tid < 64) { bs12[tid] = ffn_b1[tid] + ffn_b2[tid]; b3s[tid] = ffn_b3[tid]; }
  __syncthreads();

  // wave geometry
  const int l  = tid & 31;
  const int w  = tid >> 5;
  const int rt = w >> 1;                       // row tile 0..3
  const int nh = w & 1;                        // phaseA/B col half; stage2 K half
  const int g  = l >> 4;                       // lane K-group
  const int r  = rt * 16 + (l & 15);           // this lane's A-row
  const int nc0 = nh * 32 + (l & 15);          // phase A/B column, tile 0
  const int nc1 = nc0 + 16;                    // phase A/B column, tile 1
  const int m15 = l & 15;

  const v8f vzero = {0, 0, 0, 0, 0, 0, 0, 0};

  // ---- Phase A (WMMA): g = x@W1 + s@W2, then elu(+bias) -> as_ ----
  {
    v8f g0 = vzero, g1 = vzero;
#pragma unroll
    for (int kh = 0; kh < 2; ++kh) {
      v16h ax  = pack_a(&xs [r   * LDF + kh * 32 + 8  * g]);
      v16h bw0 = pack_b(&w1t[nc0 * LDF + kh * 32 + 8  * g]);
      v16h bw1 = pack_b(&w1t[nc1 * LDF + kh * 32 + 8  * g]);
      g0 = WMMA_F16(ax, bw0, g0);
      g1 = WMMA_F16(ax, bw1, g1);
    }
    {
      v16h av  = pack_a(&ss [r   * LDE + 8  * g]);
      v16h bw0 = pack_b(&w2t[nc0 * LDE + 8  * g]);
      v16h bw1 = pack_b(&w2t[nc1 * LDE + 8  * g]);
      g0 = WMMA_F16(av, bw0, g0);
      g1 = WMMA_F16(av, bw1, g1);
    }
    float bsum0 = bs12[nc0], bsum1 = bs12[nc1];
#pragma unroll
    for (int j = 0; j < 8; ++j) {
      int row = rt * 16 + j + 8 * g;           // C layout: M = j + 8*laneGroup
      as_[row * LDF + nc0] = eluf(g0[j] + bsum0);
      as_[row * LDF + nc1] = eluf(g1[j] + bsum1);
    }
  }
  __syncthreads();

  // ---- Phase B (WMMA): logits = elu(g)@W3 + b3 -> wts ----
  {
    v8f lg0 = vzero, lg1 = vzero;
#pragma unroll
    for (int kh = 0; kh < 2; ++kh) {
      v16h aa  = pack_a(&as_[r   * LDF + kh * 32 + 8  * g]);
      v16h bb0 = pack_b(&w3t[nc0 * LDF + kh * 32 + 8  * g]);
      v16h bb1 = pack_b(&w3t[nc1 * LDF + kh * 32 + 8  * g]);
      lg0 = WMMA_F16(aa, bb0, lg0);
      lg1 = WMMA_F16(aa, bb1, lg1);
    }
    float b30 = b3s[nc0], b31 = b3s[nc1];
#pragma unroll
    for (int j = 0; j < 8; ++j) {
      int row = rt * 16 + j + 8 * g;
      wts[row * LDF + nc0] = lg0[j] + b30;
      wts[row * LDF + nc1] = lg1[j] + b31;
    }
  }
  __syncthreads();

  // ---- softmax over f, one thread per row ----
  if (tid < 64) {
    int rr = tid;
    float m = -3.4e38f;
    for (int f = 0; f < 64; ++f) m = fmaxf(m, wts[rr * LDF + f]);
    float sum = 0.f;
    for (int f = 0; f < 64; ++f) {
      float ev = __expf(wts[rr * LDF + f] - m);
      wts[rr * LDF + f] = ev;
      sum += ev;
    }
    float inv = 1.0f / sum;
    for (int f = 0; f < 64; ++f) wts[rr * LDF + f] *= inv;
  }
  __syncthreads();

  // ---- Stage 2 (WMMA): out = [hw | w_t] x [exp_w3 ; exp_b3]
  //      Wave pair splits K: nh=0 -> fb 0..31 ; nh=1 -> fb 32..63 + bias block.
  //      Each wave covers 16 rows x all 64 columns (4 accumulators).
  v8f c0 = vzero, c1 = vzero, c2 = vzero, c3 = vzero;

  const int fb0 = nh * 32;
  const float* xrow = &xs [r * LDF];
  const float* wrow = &wts[r * LDF];

  for (int fq = 0; fq < 8; ++fq) {
    float4 xv4 = *(const float4*)(xrow + fb0 + fq * 4);
    float4 wt4 = *(const float4*)(wrow + fb0 + fq * 4);
    float xa[4] = {xv4.x, xv4.y, xv4.z, xv4.w};
    float wa[4] = {wt4.x, wt4.y, wt4.z, wt4.w};
#pragma unroll
    for (int q = 0; q < 4; ++q) {
      int fb = fb0 + fq * 4 + q;
      const _Float16* bbase = Bws + (size_t)fb * 4096 + (size_t)l * 16;
      __builtin_prefetch((const void*)(bbase + 4096), 0, 1);

      v16h a0 = make_hw(&ew1[fb * LDF + 8 * g],      &eb1[fb * LDF + 8 * g],
                        xa[q], wa[q]);
      v16h a1 = make_hw(&ew1[fb * LDF + 32 + 8 * g], &eb1[fb * LDF + 32 + 8 * g],
                        xa[q], wa[q]);

      c0 = WMMA_F16(a0, *(const v16h*)(bbase + 0 * 512), c0);
      c0 = WMMA_F16(a1, *(const v16h*)(bbase + 1 * 512), c0);
      c1 = WMMA_F16(a0, *(const v16h*)(bbase + 2 * 512), c1);
      c1 = WMMA_F16(a1, *(const v16h*)(bbase + 3 * 512), c1);
      c2 = WMMA_F16(a0, *(const v16h*)(bbase + 4 * 512), c2);
      c2 = WMMA_F16(a1, *(const v16h*)(bbase + 5 * 512), c2);
      c3 = WMMA_F16(a0, *(const v16h*)(bbase + 6 * 512), c3);
      c3 = WMMA_F16(a1, *(const v16h*)(bbase + 7 * 512), c3);
    }
  }
  if (nh == 1) {   // wave-uniform branch: EXEC stays all-ones for WMMA
    // bias K-block: A[row, f] = w_t[row, f]
    v16h a0 = pack_a(&wts[r * LDF + 8 * g]);
    v16h a1 = pack_a(&wts[r * LDF + 32 + 8 * g]);
    const _Float16* bbase = Bws + (size_t)64 * 4096 + (size_t)l * 16;
    c0 = WMMA_F16(a0, *(const v16h*)(bbase + 0 * 512), c0);
    c0 = WMMA_F16(a1, *(const v16h*)(bbase + 1 * 512), c0);
    c1 = WMMA_F16(a0, *(const v16h*)(bbase + 2 * 512), c1);
    c1 = WMMA_F16(a1, *(const v16h*)(bbase + 3 * 512), c1);
    c2 = WMMA_F16(a0, *(const v16h*)(bbase + 4 * 512), c2);
    c2 = WMMA_F16(a1, *(const v16h*)(bbase + 5 * 512), c2);
    c3 = WMMA_F16(a0, *(const v16h*)(bbase + 6 * 512), c3);
    c3 = WMMA_F16(a1, *(const v16h*)(bbase + 7 * 512), c3);
  }

  // ---- cross-wave-pair K-reduction via LDS (as_ is dead now) ----
  if (nh == 0) {
#pragma unroll
    for (int j = 0; j < 8; ++j) {
      int row = rt * 16 + j + 8 * g;
      as_[row * LDF +  0 + m15] = c0[j];
      as_[row * LDF + 16 + m15] = c1[j];
      as_[row * LDF + 32 + m15] = c2[j];
      as_[row * LDF + 48 + m15] = c3[j];
    }
  }
  __syncthreads();
  if (nh == 1) {
#pragma unroll
    for (int j = 0; j < 8; ++j) {
      int row = rt * 16 + j + 8 * g;
      size_t o = (size_t)(row_base + row) * 64;
      out[o +  0 + m15] = c0[j] + as_[row * LDF +  0 + m15];
      out[o + 16 + m15] = c1[j] + as_[row * LDF + 16 + m15];
      out[o + 32 + m15] = c2[j] + as_[row * LDF + 32 + m15];
      out[o + 48 + m15] = c3[j] + as_[row * LDF + 48 + m15];
    }
  }
}

// ---------------------------------------------------------------------------
extern "C" void kernel_launch(void* const* d_in, const int* in_sizes, int n_in,
                              void* d_out, int out_size, void* d_ws, size_t ws_size,
                              hipStream_t stream) {
  const float* x      = (const float*)d_in[0];
  const float* s      = (const float*)d_in[1];
  const float* ffn_w1 = (const float*)d_in[2];
  const float* ffn_b1 = (const float*)d_in[3];
  const float* ffn_w2 = (const float*)d_in[4];
  const float* ffn_b2 = (const float*)d_in[5];
  const float* ffn_w3 = (const float*)d_in[6];
  const float* ffn_b3 = (const float*)d_in[7];
  const float* exp_w1 = (const float*)d_in[8];
  const float* exp_b1 = (const float*)d_in[9];
  const float* exp_w3 = (const float*)d_in[10];
  const float* exp_b3 = (const float*)d_in[11];
  float* out          = (float*)d_out;
  _Float16* Bws       = (_Float16*)d_ws;   // needs 65*4096*2 = 520 KB

  vsn_prep_B<<<(NBLK * 4096 + 255) / 256, 256, 0, stream>>>(exp_w3, exp_b3, Bws);
  vsn_main<<<BT / TM, 256, 0, stream>>>(x, s, ffn_w1, ffn_b1, ffn_w2, ffn_b2,
                                        ffn_w3, ffn_b3, exp_w1, exp_b1, Bws, out);
}
